// GenerativeGraph_79499844649511
// MI455X (gfx1250) — compile-verified, matrix operand order checked by reading, not run on hardware
//
#include <hip/hip_runtime.h>

// ---------------------------------------------------------------------------
// Fused per-sample GAT-on-generated-graph kernel for gfx1250 (MI455X).
// One workgroup (256 threads = 8 wave32) per batch sample; whole per-sample
// working set lives in WGP LDS; all GEMMs via v_wmma_f32_16x16x32_bf16.
// K-accumulation chains are split/interleaved so dependent WMMAs sit >=3
// independent WMMAs apart (bf16 WMMA RAW hazard = 1 NOP + 4 coexec).
// ---------------------------------------------------------------------------

typedef __attribute__((ext_vector_type(16))) __bf16 v16bf;
typedef __attribute__((ext_vector_type(8)))  float  v8f;

#define NB 64
#define NN 512
#define DD 128
#define SLOPE 0.2f

struct alignas(16) B8 { __bf16 v[8]; };
struct alignas(8)  B4 { __bf16 v[4]; };

// ---- LDS layout (bytes) ----------------------------------------------------
constexpr int LDS_E    = 0;                  // E_s  : 512x128 bf16 = 131072
constexpr int LDS_ZT   = LDS_E    + 131072;  // ZT   : 128x512 bf16 = 131072 (z transposed)
constexpr int LDS_WS   = LDS_ZT   + 131072;  // Ws   : 128x128 bf16 =  32768 (W transposed)
constexpr int LDS_P    = LDS_WS   + 32768;   // Pbuf : 8 waves x 16x32 bf16 = 8192
constexpr int LDS_EL   = LDS_P    + 8192;    // el   : 512 f32 = 2048
constexpr int LDS_ER   = LDS_EL   + 2048;    // er   : 512 f32 = 2048
constexpr int LDS_SCOL = LDS_ER   + 2048;    // col sums: 128 f32 = 512
constexpr int LDS_OACC = LDS_SCOL + 512;     // out acc : 128 f32 = 512
constexpr int LDS_BIAS = LDS_OACC + 512;     // bias    : 128 f32 = 512
constexpr int LDS_AL   = LDS_BIAS + 512;     // attn_l  : 128 f32 = 512
constexpr int LDS_AR   = LDS_AL   + 512;     // attn_r  : 128 f32 = 512
constexpr int LDS_RED  = LDS_AR   + 512;     // reduction scratch: 256 f32 = 1024
constexpr int LDS_SCAL = LDS_RED  + 1024;    // scalars (thresh, max_el)
constexpr int LDS_TOTAL= LDS_SCAL + 64;      // = 310848 < 320 KB WGP LDS

// Fragment loader: 16x32 bf16 tile, row-major source with row stride `ld`.
// Lane L holds row (r0 + L&15), 16 contiguous elements at col c0 + (L>>4)*16.
__device__ __forceinline__ v16bf ldfrag(const __bf16* base, int ld, int r0,
                                        int c0, int lane) {
  const __bf16* p = base + (size_t)(r0 + (lane & 15)) * ld + c0 + ((lane >> 4) << 4);
  return *(const v16bf*)p;
}

__device__ __forceinline__ v8f wmma_bf16(v16bf a, v16bf b, v8f c) {
  // D(f32 16x16) = A(16x32 bf16) * B(32x16 bf16) + C
  return __builtin_amdgcn_wmma_f32_16x16x32_bf16(false, a, false, b,
                                                 (short)0, c, false, false);
}

__global__ __launch_bounds__(256, 1)
void gat_fused_kernel(const float* __restrict__ emb, const float* __restrict__ Wg,
                      const float* __restrict__ attn_l, const float* __restrict__ attn_r,
                      const float* __restrict__ bias, float* __restrict__ out) {
  extern __shared__ __align__(128) char smem[];
  __bf16* E_s   = (__bf16*)(smem + LDS_E);
  __bf16* ZT    = (__bf16*)(smem + LDS_ZT);
  __bf16* Ws    = (__bf16*)(smem + LDS_WS);
  __bf16* Pbuf  = (__bf16*)(smem + LDS_P);
  float*  el_s  = (float*)(smem + LDS_EL);
  float*  er_s  = (float*)(smem + LDS_ER);
  float*  s_col = (float*)(smem + LDS_SCOL);
  float*  o_acc = (float*)(smem + LDS_OACC);
  float*  bias_s= (float*)(smem + LDS_BIAS);
  float*  al_s  = (float*)(smem + LDS_AL);
  float*  ar_s  = (float*)(smem + LDS_AR);
  float*  red   = (float*)(smem + LDS_RED);
  float*  scal  = (float*)(smem + LDS_SCAL);

  const int tid  = threadIdx.x;
  const int lane = tid & 31;        // wave32 lane
  const int w    = tid >> 5;        // wave id (0..7)
  const int b    = blockIdx.x;      // sample

  // ---- init ---------------------------------------------------------------
  if (tid < DD) { s_col[tid] = 0.f; o_acc[tid] = 0.f; }
  __syncthreads();

  // ---- phase 0: load E -> bf16 LDS (row-major) + column sums for thresh ---
  // thresh = mean(E E^T) = |sum_i E_i|^2 / N^2  (closed form, no score pass)
  {
    const float4* E4 = (const float4*)(emb + (size_t)b * NN * DD);
    const int kb = (tid * 4) & (DD - 1);       // fixed k-base for this thread
    float cs0 = 0.f, cs1 = 0.f, cs2 = 0.f, cs3 = 0.f;
    #pragma unroll 4
    for (int m = 0; m < (NN * DD / 4) / 256; ++m) {   // 64 iters
      int g = tid + 256 * m;
      float4 v = E4[g];
      cs0 += v.x; cs1 += v.y; cs2 += v.z; cs3 += v.w;
      B4 pk;
      pk.v[0] = (__bf16)v.x; pk.v[1] = (__bf16)v.y;
      pk.v[2] = (__bf16)v.z; pk.v[3] = (__bf16)v.w;
      *(B4*)(E_s + (size_t)g * 4) = pk;               // ds_store_b64
    }
    atomicAdd(&s_col[kb + 0], cs0);
    atomicAdd(&s_col[kb + 1], cs1);
    atomicAdd(&s_col[kb + 2], cs2);
    atomicAdd(&s_col[kb + 3], cs3);

    // W transpose into Ws[d][k] (K contiguous per output column for B-frags)
    #pragma unroll 4
    for (int m = 0; m < (DD * DD) / 256; ++m) {       // 64 iters
      int idx = tid + 256 * m;
      int k = idx >> 7, d = idx & (DD - 1);
      Ws[d * DD + k] = (__bf16)Wg[idx];
    }
    if (tid < DD) {
      bias_s[tid] = bias[tid];
      al_s[tid]   = attn_l[tid];
      ar_s[tid]   = attn_r[tid];
    }
  }
  __syncthreads();

  // ---- thresh reduction ---------------------------------------------------
  if (tid < DD) { float s = s_col[tid]; red[tid] = s * s; }
  __syncthreads();
  for (int off = 64; off > 0; off >>= 1) {
    if (tid < off) red[tid] += red[tid + off];
    __syncthreads();
  }
  if (tid == 0) scal[0] = red[0] * (1.f / ((float)NN * (float)NN));

  // ---- phase 1: z = E * W, stored transposed as ZT[d][n] (bf16) -----------
  // Two d-tiles at once, two K-chains each: 4 independent accumulators,
  // dependent WMMAs are 3 apart -> no hazard nops.
  for (int tn = w; tn < NN / 16; tn += 8) {
    #pragma unroll 1
    for (int td = 0; td < DD / 16; td += 2) {
      v8f cA0 = {}, cA1 = {}, cB0 = {}, cB1 = {};
      #pragma unroll
      for (int kk = 0; kk < 4; ++kk) {
        int k0 = kk * 32;
        v16bf a   = ldfrag(E_s, DD, tn * 16, k0, lane);
        v16bf bbA = ldfrag(Ws,  DD, td * 16, k0, lane);
        v16bf bbB = ldfrag(Ws,  DD, (td + 1) * 16, k0, lane);
        if ((kk & 1) == 0) { cA0 = wmma_bf16(a, bbA, cA0); cB0 = wmma_bf16(a, bbB, cB0); }
        else               { cA1 = wmma_bf16(a, bbA, cA1); cB1 = wmma_bf16(a, bbB, cB1); }
      }
      v8f cA = cA0 + cA1;
      v8f cB = cB0 + cB1;
      // C layout: lane holds (n = tn*16 + r + 8*(lane>>4), d = td*16 + lane&15)
      B8 pkA, pkB;
      #pragma unroll
      for (int r = 0; r < 8; ++r) { pkA.v[r] = (__bf16)cA[r]; pkB.v[r] = (__bf16)cB[r]; }
      int dA = td * 16 + (lane & 15);
      int n0 = tn * 16 + ((lane >> 4) << 3);
      *(B8*)(ZT + (size_t)dA * NN + n0) = pkA;          // ds_store_b128
      *(B8*)(ZT + (size_t)(dA + 16) * NN + n0) = pkB;
    }
  }
  __syncthreads();

  // ---- phase 2: el/er per node + max(el) ----------------------------------
  float lmax = -1e30f;
  for (int n = tid; n < NN; n += 256) {
    float accl = 0.f, accr = 0.f;
    #pragma unroll 8
    for (int d = 0; d < DD; ++d) {
      float zv = (float)ZT[(size_t)d * NN + n];
      accl += zv * al_s[d];
      accr += zv * ar_s[d];
    }
    el_s[n] = accl;
    er_s[n] = accr;
    lmax = fmaxf(lmax, accl);
  }
  red[tid] = lmax;
  __syncthreads();
  for (int off = 128; off > 0; off >>= 1) {
    if (tid < off) red[tid] = fmaxf(red[tid], red[tid + off]);
    __syncthreads();
  }
  if (tid == 0) scal[1] = red[0];
  __syncthreads();

  const float thresh = scal[0];
  const float mx_el  = scal[1];

  // ---- phase 3: streaming attention, per 16-dst-column tile per wave ------
  __bf16* Pw = Pbuf + w * (16 * 32);                  // wave-private P tile
  #pragma unroll 1
  for (int jt = w; jt < NN / 16; jt += 8) {
    const int j0    = jt * 16;
    const int jlane = j0 + (lane & 15);
    const float erj = er_s[jlane];
    float mjt = mx_el + erj;                          // exact column max bound
    const float mj = (mjt > 0.f) ? mjt : SLOPE * mjt; // lrelu is monotone

    v8f H[8];
    #pragma unroll
    for (int dt = 0; dt < 8; ++dt) H[dt] = v8f{};
    float dsum = 0.f;

    // hoist the 4 K-fragments of the dst rows (B operand of E*E^T)
    v16bf jb[4];
    #pragma unroll
    for (int kk = 0; kk < 4; ++kk)
      jb[kk] = ldfrag(E_s, DD, j0, kk * 32, lane);

    #pragma unroll 1
    for (int ib = 0; ib < NN; ib += 32) {             // stream 32 src rows
      // scores for both 16-row halves, interleaved 2x2 accumulator chains
      v8f c00 = {}, c01 = {}, c10 = {}, c11 = {};
      #pragma unroll
      for (int kk = 0; kk < 4; ++kk) {
        int k0 = kk * 32;
        v16bf a0 = ldfrag(E_s, DD, ib,      k0, lane);
        v16bf a1 = ldfrag(E_s, DD, ib + 16, k0, lane);
        if ((kk & 1) == 0) { c00 = wmma_bf16(a0, jb[kk], c00); c10 = wmma_bf16(a1, jb[kk], c10); }
        else               { c01 = wmma_bf16(a0, jb[kk], c01); c11 = wmma_bf16(a1, jb[kk], c11); }
      }
      v8f sc[2];
      sc[0] = c00 + c01;
      sc[1] = c10 + c11;

      // el for this lane's 8 rows per half: vectorized ds_load_b128 pairs
      const int rb = ib + ((lane >> 4) << 3);
      float ev0[8], ev1[8];
      *(float4*)&ev0[0] = *(const float4*)(el_s + rb);
      *(float4*)&ev0[4] = *(const float4*)(el_s + rb + 4);
      *(float4*)&ev1[0] = *(const float4*)(el_s + rb + 16);
      *(float4*)&ev1[4] = *(const float4*)(el_s + rb + 20);

      // P[i][j] = mask ? exp(lrelu(el_i+er_j) - m_j) : 0
      #pragma unroll
      for (int ti = 0; ti < 2; ++ti) {
        B8 pk;
        #pragma unroll
        for (int r = 0; r < 8; ++r) {
          int i = rb + ti * 16 + r;
          bool m = (sc[ti][r] > thresh) | (i == jlane); // threshold | self-loop
          float e = (ti ? ev1[r] : ev0[r]) + erj;
          e = (e > 0.f) ? e : SLOPE * e;
          float p = m ? __expf(e - mj) : 0.f;
          dsum += p;
          pk.v[r] = (__bf16)p;
        }
        // stash P^T-ready: Pw[j_local][i_local], i contiguous
        *(B8*)(Pw + (lane & 15) * 32 + ti * 16 + ((lane >> 4) << 3)) = pk;
      }

      // H^T[d][j] += Z^T[d][i] * P[i][j]  (K = 32 src rows this step);
      // 8 distinct accumulators -> dependency distance 7, no nops.
      v16bf pb = ldfrag(Pw, 32, 0, 0, lane);          // same wave: LDS in-order
      #pragma unroll
      for (int dt = 0; dt < 8; ++dt) {
        v16bf a = ldfrag(ZT, NN, dt * 16, ib, lane);
        H[dt] = wmma_bf16(a, pb, H[dt]);
      }
    }

    // denominator: fold lane <-> lane^16 partials (same j column)
    float denom = dsum + __shfl_xor(dsum, 16, 32);
    const float rden = 1.f / fmaxf(denom, 1e-20f);

    // epilogue: elu(H*rden + bias), accumulate node-mean
    #pragma unroll
    for (int dt = 0; dt < 8; ++dt) {
      #pragma unroll
      for (int r = 0; r < 8; ++r) {
        int d = dt * 16 + r + ((lane >> 4) << 3);
        float v = H[dt][r] * rden + bias_s[d];
        float ve = (v > 0.f) ? v : (__expf(v) - 1.f);
        atomicAdd(&o_acc[d], ve);                     // ds_add_f32
      }
    }
  }

  __syncthreads();
  if (tid < DD) out[(size_t)b * DD + tid] = o_acc[tid] * (1.f / (float)NN);
}

extern "C" void kernel_launch(void* const* d_in, const int* in_sizes, int n_in,
                              void* d_out, int out_size, void* d_ws, size_t ws_size,
                              hipStream_t stream) {
  (void)in_sizes; (void)n_in; (void)d_ws; (void)ws_size; (void)out_size;
  const float* emb    = (const float*)d_in[0];
  const float* Wg     = (const float*)d_in[1];
  const float* attn_l = (const float*)d_in[2];
  const float* attn_r = (const float*)d_in[3];
  const float* bias   = (const float*)d_in[4];
  float* out          = (float*)d_out;

  // 310848 B dynamic LDS (< 320 KB WGP limit) — raise the cap, every call
  // (deterministic, no static guards).
  (void)hipFuncSetAttribute((const void*)gat_fused_kernel,
                            hipFuncAttributeMaxDynamicSharedMemorySize,
                            LDS_TOTAL);
  gat_fused_kernel<<<dim3(NB), dim3(256), LDS_TOTAL, stream>>>(
      emb, Wg, attn_l, attn_r, bias, out);
}
